// Attention_69861938037658
// MI455X (gfx1250) — compile-verified, matrix operand order checked by reading, not run on hardware
//
#include <hip/hip_runtime.h>
#include <math.h>

typedef __attribute__((ext_vector_type(16))) _Float16 v16h;
typedef __attribute__((ext_vector_type(8)))  _Float16 v8h;
typedef __attribute__((ext_vector_type(4)))  _Float16 v4h;
typedef __attribute__((ext_vector_type(8)))  float    v8f;

#define BATCH 4096
#define S 200
#define SP 208          // padded to 13 M-tiles of 16
#define H 128
#define NEGV -1.0e9f

// LDS layout (bytes):
//   Ah   : SP*H f16              = 53248
//   W1h  : H*H f16               = 32768
//   plast: H f32                 =   512
//   w3   : H f32                 =   512
//   scores: 256 f32              =  1024
//   red  : 256 f32               =  1024
#define LDS_BYTES (53248 + 32768 + 512 + 512 + 1024 + 1024)

__device__ __forceinline__ float fast_tanh(float x) {
#if __has_builtin(__builtin_amdgcn_tanhf)
    // CDNA5 hardware transcendental: single v_tanh_f32 (TRANS pipe,
    // co-executes with WMMA which is also TRANS-tracked on gfx1250).
    return __builtin_amdgcn_tanhf(x);
#else
    x = fminf(15.f, fmaxf(-15.f, x));
    float e2 = __expf(2.f * x);
    // tanh = 1 - 2/(e2+1); v_rcp_f32 instead of exact IEEE division
    return fmaf(-2.f, __builtin_amdgcn_rcpf(e2 + 1.f), 1.f);
#endif
}

extern "C" __global__ __launch_bounds__(256, 1)
void attn_fused_kernel(const float* __restrict__ all_memory,
                       const float* __restrict__ last_memory,
                       const int*   __restrict__ mask,
                       const float* __restrict__ W1,
                       const float* __restrict__ W2,
                       const float* __restrict__ W3_w,
                       const float* __restrict__ W3_b,
                       float* __restrict__ out)
{
    extern __shared__ char lds_raw[];
    _Float16* Ah     = (_Float16*)lds_raw;                    // [SP][H]
    _Float16* W1h    = (_Float16*)(lds_raw + 53248);          // [H][H] row-major [o][h]
    float*    plast  = (float*)(lds_raw + 53248 + 32768);     // [H]
    float*    w3     = plast + H;                             // [H]
    float*    scores = w3 + H;                                // [256]
    float*    red    = scores + 256;                          // [256]

    const int b    = blockIdx.x;
    const int t    = threadIdx.x;
    const int lane = t & 31;
    const int wave = t >> 5;
    const int lh     = lane & 15;   // column (B/C/D) or row (A) within tile
    const int halfid = lane >> 4;   // K-half selector / row-offset selector

    // ---------------- Phase 1: stage A and W1 into LDS as f16 ----------------
    {
        const float4* Ag = (const float4*)(all_memory + (size_t)b * S * H);
        v4h* Ah4 = (v4h*)Ah;
        for (int i = t; i < (S * H) / 4; i += 256) {
            float4 v = Ag[i];
            v4h hv;
            hv[0] = (_Float16)v.x; hv[1] = (_Float16)v.y;
            hv[2] = (_Float16)v.z; hv[3] = (_Float16)v.w;
            Ah4[i] = hv;
        }
        // zero-pad rows 200..207 (8*128/4 = 256 v4h chunks)
        {
            v4h z; z[0] = z[1] = z[2] = z[3] = (_Float16)0.f;
            Ah4[(S * H) / 4 + t] = z;
        }
        const float4* W1g = (const float4*)W1;
        v4h* W1h4 = (v4h*)W1h;
        for (int i = t; i < (H * H) / 4; i += 256) {
            float4 v = W1g[i];
            v4h hv;
            hv[0] = (_Float16)v.x; hv[1] = (_Float16)v.y;
            hv[2] = (_Float16)v.z; hv[3] = (_Float16)v.w;
            W1h4[i] = hv;
        }
        // proj_last[o] = sum_h last[b,h] * W2[o,h]  (f32, trivial cost)
        if (t < H) {
            const float4* lm  = (const float4*)(last_memory + (size_t)b * H);
            const float4* w2r = (const float4*)(W2 + (size_t)t * H);
            float acc = 0.f;
            #pragma unroll 4
            for (int i = 0; i < H / 4; ++i) {
                float4 a = lm[i], w = w2r[i];
                acc += a.x * w.x + a.y * w.y + a.z * w.z + a.w * w.w;
            }
            plast[t] = acc;
            w3[t]    = W3_w[t];
        }
    }
    __syncthreads();

    // ---------------- Phase 2: WMMA GEMM + fused tanh + W3 dot ----------------
    const float w3b = W3_b[0];
    for (int m = wave; m < 13; m += 8) {
        // Preload the 4 K-fragments of A for this M-tile.
        // A 16-bit 16x32 frag: lane(0-15)=row M, halves: K[kh..kh+7] and K[kh+16..kh+23], kh=(lane>>4)*8
        v16h afrag[4];
        const _Float16* arow = Ah + (size_t)(m * 16 + lh) * H + halfid * 8;
        #pragma unroll
        for (int k = 0; k < 4; ++k) {
            union { v16h v; v8h h[2]; } u;
            u.h[0] = *(const v8h*)(arow + k * 32);
            u.h[1] = *(const v8h*)(arow + k * 32 + 16);
            afrag[k] = u.v;
        }
        float part[8];
        #pragma unroll
        for (int r = 0; r < 8; ++r) part[r] = 0.f;

        #pragma unroll 2
        for (int n = 0; n < 8; ++n) {
            // B 32x16 frag = W1^T tile: lane holds column N=o=n*16+lh, i.e. row o of W1 over K
            const _Float16* brow = W1h + (size_t)(n * 16 + lh) * H + halfid * 8;
            v8f acc = {};
            #pragma unroll
            for (int k = 0; k < 4; ++k) {
                union { v16h v; v8h h[2]; } u;
                u.h[0] = *(const v8h*)(brow + k * 32);
                u.h[1] = *(const v8h*)(brow + k * 32 + 16);
                acc = __builtin_amdgcn_wmma_f32_16x16x32_f16(
                        false, afrag[k], false, u.v, (short)0, acc, false, false);
            }
            // Fused epilogue: hidden = tanh(P + plast[o]); score += hidden * w3[o]
            const int   col = n * 16 + lh;
            const float pl  = plast[col];
            const float wv  = w3[col];
            #pragma unroll
            for (int r = 0; r < 8; ++r)
                part[r] += fast_tanh(acc[r] + pl) * wv;
        }
        // Reduce over the 16 lanes of each half-wave (columns of this lane group)
        #pragma unroll
        for (int r = 0; r < 8; ++r) {
            float v = part[r];
            v += __shfl_xor(v, 1, 16);
            v += __shfl_xor(v, 2, 16);
            v += __shfl_xor(v, 4, 16);
            v += __shfl_xor(v, 8, 16);
            part[r] = v;
        }
        if (lh == 0) {
            const int base = m * 16 + halfid * 8;   // lanes 0 / 16 hold rows base..base+7
            #pragma unroll
            for (int r = 0; r < 8; ++r) {
                int s = base + r;
                if (s < S) scores[s] = part[r] + w3b;
            }
        }
    }
    __syncthreads();

    // ---------------- Phase 3: masked softmax over S ----------------
    const int* maskb = mask + (size_t)b * S;
    float sval = -INFINITY;
    if (t < S) sval = maskb[t] ? NEGV : scores[t];
    red[t] = sval;
    __syncthreads();
    for (int off = 128; off > 0; off >>= 1) {
        if (t < off) red[t] = fmaxf(red[t], red[t + off]);
        __syncthreads();
    }
    const float mx = red[0];
    __syncthreads();
    const float e = (t < S) ? __expf(sval - mx) : 0.f;
    red[t] = e;
    __syncthreads();
    for (int off = 128; off > 0; off >>= 1) {
        if (t < off) red[t] += red[t + off];
        __syncthreads();
    }
    const float inv_denom = __builtin_amdgcn_rcpf(red[0]);
    if (t < S) scores[t] = e * inv_denom;   // attn weights
    __syncthreads();

    // ---------------- Phase 4: out[b,h] = sum_s attn[s] * A[s,h] ----------------
    if (t < H) {
        float acc = 0.f;
        const _Float16* colp = Ah + t;
        #pragma unroll 8
        for (int s = 0; s < S; ++s)
            acc += scores[s] * (float)colp[(size_t)s * H];
        out[(size_t)b * H + t] = acc;
    }
}

extern "C" void kernel_launch(void* const* d_in, const int* in_sizes, int n_in,
                              void* d_out, int out_size, void* d_ws, size_t ws_size,
                              hipStream_t stream) {
    const float* all_memory  = (const float*)d_in[0];
    const float* last_memory = (const float*)d_in[1];
    const int*   mask        = (const int*)d_in[2];
    const float* W1          = (const float*)d_in[3];
    const float* W2          = (const float*)d_in[4];
    const float* W3_w        = (const float*)d_in[5];
    const float* W3_b        = (const float*)d_in[6];
    float* out = (float*)d_out;

    (void)in_sizes; (void)n_in; (void)out_size; (void)d_ws; (void)ws_size;

    hipFuncSetAttribute((const void*)attn_fused_kernel,
                        hipFuncAttributeMaxDynamicSharedMemorySize, LDS_BYTES);

    attn_fused_kernel<<<BATCH, 256, LDS_BYTES, stream>>>(
        all_memory, last_memory, mask, W1, W2, W3_w, W3_b, out);
}
